// ConditionalCrossAttnDiT_19215683682459
// MI455X (gfx1250) — compile-verified, matrix operand order checked by reading, not run on hardware
//
#include <hip/hip_runtime.h>
#include <math.h>

#define BB 32
#define DD 384
#define TT 256
#define HH 6
#define MMH 1536
#define PD 768
#define NTOK (BB*TT)          // 8192
#define QKVLD 1152

typedef __attribute__((ext_vector_type(16))) __bf16 v16bf;
typedef __attribute__((ext_vector_type(8)))  __bf16 v8bf;
typedef __attribute__((ext_vector_type(8)))  float  v8f;

// ---------- helpers ----------
__device__ __forceinline__ __bf16 f2bf(float f) {
  unsigned int u = __builtin_bit_cast(unsigned int, f);
  unsigned int r = (u + 0x7FFFu + ((u >> 16) & 1u)) >> 16;
  return __builtin_bit_cast(__bf16, (unsigned short)r);
}

// bf16 fragment, contiguous row-major source (2x global/ds _b128 per lane):
// lane&15 = row, lane>>4 = K-half. x[0..7] = p[hl*8 + i], x[8..15] = p[16 + hl*8 + i].
__device__ __forceinline__ v16bf load_frag_bf(const __bf16* p, int hl) {
  v8bf a = *(const v8bf*)(p + hl * 8);
  v8bf b = *(const v8bf*)(p + 16 + hl * 8);
  return __builtin_shufflevector(a, b, 0,1,2,3,4,5,6,7,8,9,10,11,12,13,14,15);
}

// fp32 source fragment (LDS scores): convert on load
__device__ __forceinline__ v16bf load_frag_f32(const float* p, int hl) {
  v16bf x;
#pragma unroll
  for (int i = 0; i < 8; ++i) {
    x[i]     = f2bf(p[hl * 8 + i]);
    x[8 + i] = f2bf(p[16 + hl * 8 + i]);
  }
  return x;
}

__device__ __forceinline__ v8f wmma_bf16(v16bf a, v16bf b, v8f c) {
  return __builtin_amdgcn_wmma_f32_16x16x32_bf16(false, a, false, b, (short)0, c, false, false);
}

// ---------- generic GEMM: out = act(A @ W^T + bias) [+ res] ----------
// A: N x K bf16 (row-major). W: M x K bf16 (row-major). bias/res fp32.
// Each wave computes 16 rows x 64 cols. ACT: 0 none, 1 silu, 2 gelu. OBF: 1 -> bf16 out.
template<int ACT, int OBF>
__global__ __launch_bounds__(256)
void gemm_wmma_kernel(const __bf16* __restrict__ A, const __bf16* __restrict__ W,
                      const float* __restrict__ bias, const float* res, int ldres,
                      void* outp, int ldc, int N, int K, int M)
{
  int wave = (blockIdx.x * blockDim.x + threadIdx.x) >> 5;
  int lane = threadIdx.x & 31;
  int tilesM = M >> 6;
  int tilesN = N >> 4;
  int tileN = wave / tilesM;
  int tileM = wave - tileN * tilesM;
  if (tileN >= tilesN) return;          // wave-uniform

  int i0 = tileN << 4;
  int j0 = tileM << 6;
  int r  = lane & 15;
  int hl = lane >> 4;

  const __bf16* Arow = A + (size_t)(i0 + r) * K;
  const __bf16* Wrow = W + (size_t)(j0 + r) * K;

  v8f acc[4] = {};
  for (int kk = 0; kk < K; kk += 32) {
    __builtin_prefetch(Arow + kk + 64, 0, 1);       // global_prefetch_b8 (stream-ahead)
    v16bf a = load_frag_bf(Arow + kk, hl);
#pragma unroll
    for (int jt = 0; jt < 4; ++jt) {
      v16bf b = load_frag_bf(Wrow + (size_t)jt * 16 * K + kk, hl);
      acc[jt] = wmma_bf16(a, b, acc[jt]);
    }
  }

#pragma unroll
  for (int jt = 0; jt < 4; ++jt) {
#pragma unroll
    for (int v = 0; v < 8; ++v) {
      int row = i0 + v + (hl << 3);
      int col = j0 + jt * 16 + r;
      float val = acc[jt][v] + bias[col];
      if (ACT == 1) val = val / (1.0f + __expf(-val));
      else if (ACT == 2) val = 0.5f * val * (1.0f + erff(val * 0.70710678118654752f));
      if (res) val += res[(size_t)row * ldres + col];
      if (OBF) ((__bf16*)outp)[(size_t)row * ldc + col] = f2bf(val);
      else     ((float*)outp)[(size_t)row * ldc + col] = val;
    }
  }
}

// ---------- attention ----------
// qkv: NTOK x 1152 bf16, [q|k|v], head h at col h*64. o: NTOK x 384 bf16.
// grid = B*H*(T/64), 128 threads (4 waves); wave owns 16 query rows.
__global__ __launch_bounds__(128)
void attn_kernel(const __bf16* __restrict__ qkv, __bf16* __restrict__ o)
{
  __shared__ __align__(16) float  S[4][16][260];   // 260 % 64 != 0 -> conflict-free
  __shared__ __align__(16) __bf16 Vs[64][264];     // V transposed: [d][k]; 264*2B = 16B mult

  int w    = threadIdx.x >> 5;
  int lane = threadIdx.x & 31;
  int r    = lane & 15;
  int hl   = lane >> 4;

  int bh = blockIdx.x >> 2;
  int qt = blockIdx.x & 3;
  int b  = bh / HH;
  int h  = bh - b * HH;
  int q0 = qt * 64 + w * 16;

  const __bf16* base = qkv + (size_t)(b * TT) * QKVLD + h * 64;

  // phase 0: cooperative transpose copy of V (256 x 64) into LDS (shared by all waves)
  for (int c = threadIdx.x; c < 2048; c += 128) {
    int k  = c >> 3;
    int d0 = (c & 7) * 8;
    v8bf v = *(const v8bf*)(base + 2 * DD + (size_t)k * QKVLD + d0);
#pragma unroll
    for (int i = 0; i < 8; ++i) Vs[d0 + i][k] = v[i];
  }

  // phase 1: S = (Q K^T) / sqrt(64)
  const __bf16* Qrow = base + (size_t)(q0 + r) * QKVLD;
  v16bf aq0 = load_frag_bf(Qrow, hl);
  v16bf aq1 = load_frag_bf(Qrow + 32, hl);
  for (int kt = 0; kt < 16; ++kt) {
    const __bf16* Krow = base + DD + (size_t)(kt * 16 + r) * QKVLD;
    v16bf b0 = load_frag_bf(Krow, hl);
    v16bf b1 = load_frag_bf(Krow + 32, hl);
    v8f c = {};
    c = wmma_bf16(aq0, b0, c);
    c = wmma_bf16(aq1, b1, c);
#pragma unroll
    for (int v = 0; v < 8; ++v)
      S[w][v + (hl << 3)][kt * 16 + r] = c[v] * 0.125f;
  }
  __syncthreads();

  // phase 2: softmax (one lane per query row)
  if (lane < 16) {
    float* Srow = S[w][lane];
    float mx = -3.4e38f;
    for (int j = 0; j < TT; ++j) mx = fmaxf(mx, Srow[j]);
    float sum = 0.0f;
    for (int j = 0; j < TT; ++j) { float e = __expf(Srow[j] - mx); Srow[j] = e; sum += e; }
    float inv = 1.0f / sum;
    for (int j = 0; j < TT; ++j) Srow[j] *= inv;
  }
  __syncthreads();

  // phase 3: O = P @ V  (B-frags from transposed LDS V, 2x ds_load_b128 each)
  v8f acc[4] = {};
  for (int kk = 0; kk < TT; kk += 32) {
    v16bf a = load_frag_f32(&S[w][r][kk], hl);
#pragma unroll
    for (int jt = 0; jt < 4; ++jt) {
      v16bf bb = load_frag_bf(&Vs[jt * 16 + r][kk], hl);
      acc[jt] = wmma_bf16(a, bb, acc[jt]);
    }
  }
#pragma unroll
  for (int jt = 0; jt < 4; ++jt)
#pragma unroll
    for (int v = 0; v < 8; ++v)
      o[(size_t)(b * TT + q0 + v + (hl << 3)) * DD + h * 64 + jt * 16 + r] = f2bf(acc[jt][v]);
}

// ---------- adaLN: qout = bf16( (x + gate*rms(x,g)) * (1+sc) + sh ) ----------
__global__ __launch_bounds__(256)
void adaln_mod_kernel(const float* __restrict__ x, const float* __restrict__ ada,
                      const float* __restrict__ g, __bf16* __restrict__ qout)
{
  int wave = threadIdx.x >> 5;
  int lane = threadIdx.x & 31;
  int row  = blockIdx.x * 8 + wave;
  int b    = row >> 8;
  const float* xr = x + (size_t)row * DD;
  float ss = 0.0f;
#pragma unroll
  for (int i = lane; i < DD; i += 32) { float v = xr[i]; ss += v * v; }
#pragma unroll
  for (int m = 16; m >= 1; m >>= 1) ss += __shfl_xor(ss, m, 32);
  float nrm = fmaxf(sqrtf(ss), 1e-12f);
  float scl = 19.595917942265426f * g[0] / nrm;   // sqrt(384)*g / ||x||
  const float* ab = ada + (size_t)b * (3 * DD);
#pragma unroll
  for (int i = lane; i < DD; i += 32) {
    float xv = xr[i];
    float x2 = xv + ab[2 * DD + i] * (xv * scl);
    qout[(size_t)row * DD + i] = f2bf(x2 * (1.0f + ab[DD + i]) + ab[i]);
  }
}

// ---------- final modulation: out = bf16( rms(x,g)*(1+sc)+sh ) ----------
__global__ __launch_bounds__(256)
void final_mod_kernel(const float* __restrict__ x, const float* __restrict__ fm,
                      const float* __restrict__ g, __bf16* __restrict__ qout)
{
  int wave = threadIdx.x >> 5;
  int lane = threadIdx.x & 31;
  int row  = blockIdx.x * 8 + wave;
  int b    = row >> 8;
  const float* xr = x + (size_t)row * DD;
  float ss = 0.0f;
#pragma unroll
  for (int i = lane; i < DD; i += 32) { float v = xr[i]; ss += v * v; }
#pragma unroll
  for (int m = 16; m >= 1; m >>= 1) ss += __shfl_xor(ss, m, 32);
  float nrm = fmaxf(sqrtf(ss), 1e-12f);
  float scl = 19.595917942265426f * g[0] / nrm;
  const float* fb = fm + (size_t)b * (2 * DD);
#pragma unroll
  for (int i = lane; i < DD; i += 32)
    qout[(size_t)row * DD + i] = f2bf((xr[i] * scl) * (1.0f + fb[DD + i]) + fb[i]);
}

// ---------- timestep embedding (bf16 out) ----------
__global__ void tfreq_kernel(const float* __restrict__ t, __bf16* __restrict__ e)
{
  int idx = blockIdx.x * blockDim.x + threadIdx.x;
  if (idx >= BB * 256) return;
  int b = idx >> 8, j = idx & 255, jj = j & 127;
  float freq = __expf(-9.210340371976184f * (float)jj / 128.0f);
  float a = t[b] * 1000.0f * freq;
  e[idx] = f2bf((j < 128) ? cosf(a) : sinf(a));
}

__global__ void silu_bf_kernel(const float* __restrict__ in, __bf16* __restrict__ out, int n)
{
  int idx = blockIdx.x * blockDim.x + threadIdx.x;
  if (idx >= n) return;
  float v = in[idx];
  out[idx] = f2bf(v / (1.0f + __expf(-v)));
}

__global__ void f2bf_kernel(const float* __restrict__ in, __bf16* __restrict__ out, int n)
{
  int idx = blockIdx.x * blockDim.x + threadIdx.x;
  if (idx >= n) return;
  out[idx] = f2bf(in[idx]);
}

__global__ void addpos_kernel(float* __restrict__ x, const float* __restrict__ pos)
{
  int idx = blockIdx.x * blockDim.x + threadIdx.x;
  if (idx >= NTOK * DD) return;
  x[idx] += pos[idx % (TT * DD)];
}

// embed feature layout: f = c*256 + iy*16 + ix  (bf16 out)
__global__ void patch_gather_kernel(const float* __restrict__ img, __bf16* __restrict__ out)
{
  int idx = blockIdx.x * blockDim.x + threadIdx.x;
  if (idx >= NTOK * PD) return;
  int f = idx % PD, tok = idx / PD;
  int b = tok >> 8, t = tok & 255;
  int py = t >> 4, px = t & 15;
  int c = f >> 8, rem = f & 255;
  int iy = rem >> 4, ix = rem & 15;
  out[idx] = f2bf(img[(((size_t)(b * 3 + c) * 256 + py * 16 + iy) * 256) + px * 16 + ix]);
}

// final feature layout: f = iy*48 + ix*3 + c  (fp32 tokens -> fp32 image)
__global__ void unpatchify_kernel(const float* __restrict__ ftok, float* __restrict__ out)
{
  int idx = blockIdx.x * blockDim.x + threadIdx.x;
  if (idx >= BB * 3 * 256 * 256) return;
  int xc = idx & 255;
  int rest = idx >> 8;
  int yr = rest & 255; rest >>= 8;
  int c = rest % 3;
  int b = rest / 3;
  int py = yr >> 4, iy = yr & 15, px = xc >> 4, ix = xc & 15;
  out[idx] = ftok[(size_t)(b * TT + py * 16 + px) * PD + iy * 48 + ix * 3 + c];
}

// ---------- host side ----------
static void gemm(hipStream_t s, int act, int obf, const __bf16* A, const __bf16* W,
                 const float* bias, const float* res, int ldres, void* out, int ldc,
                 int N, int K, int M)
{
  int waves = (N / 16) * (M / 64);
  int blocks = (waves + 7) / 8;
  if (act == 0 && obf == 0)
    gemm_wmma_kernel<0,0><<<blocks, 256, 0, s>>>(A, W, bias, res, ldres, out, ldc, N, K, M);
  else if (act == 0 && obf == 1)
    gemm_wmma_kernel<0,1><<<blocks, 256, 0, s>>>(A, W, bias, res, ldres, out, ldc, N, K, M);
  else if (act == 1)
    gemm_wmma_kernel<1,1><<<blocks, 256, 0, s>>>(A, W, bias, res, ldres, out, ldc, N, K, M);
  else
    gemm_wmma_kernel<2,1><<<blocks, 256, 0, s>>>(A, W, bias, res, ldres, out, ldc, N, K, M);
}

extern "C" void kernel_launch(void* const* d_in, const int* in_sizes, int n_in,
                              void* d_out, int out_size, void* d_ws, size_t ws_size,
                              hipStream_t stream)
{
  (void)in_sizes; (void)n_in; (void)out_size; (void)ws_size;
  auto F = [&](int i) { return (const float*)d_in[i]; };

  const float* x_in  = F(0);
  const float* t_in  = F(1);
  const float* r_in  = F(2);
  const float* cond  = F(3);
  int p = 4;
  const float* x_emb_w = F(p++); const float* x_emb_b = F(p++);
  const float* c_emb_w = F(p++); const float* c_emb_b = F(p++);
  const float* pos_x   = F(p++); const float* pos_c   = F(p++);
  const float* t_w1 = F(p++); const float* t_b1 = F(p++);
  const float* t_w2 = F(p++); const float* t_b2 = F(p++);
  const float* r_w1 = F(p++); const float* r_b1 = F(p++);
  const float* r_w2 = F(p++); const float* r_b2 = F(p++);
  struct BP {
    const float *sa_in_w, *sa_in_b, *sa_out_w, *sa_out_b;
    const float *ca_in_w, *ca_in_b, *ca_out_w, *ca_out_b;
    const float *k_w, *k_b, *v_w, *v_b, *po_w, *po_b;
    const float *ada_self_w, *ada_self_b, *ada_self_g;
    const float *ada_cross_w, *ada_cross_b, *ada_cross_g;
    const float *ada_mlp_w, *ada_mlp_b, *ada_mlp_g;
    const float *mlp_w1, *mlp_b1, *mlp_w2, *mlp_b2;
  } bp[6];
  for (int i = 0; i < 6; ++i) {
    bp[i].sa_in_w = F(p++); bp[i].sa_in_b = F(p++); bp[i].sa_out_w = F(p++); bp[i].sa_out_b = F(p++);
    bp[i].ca_in_w = F(p++); bp[i].ca_in_b = F(p++); bp[i].ca_out_w = F(p++); bp[i].ca_out_b = F(p++);
    bp[i].k_w = F(p++); bp[i].k_b = F(p++); bp[i].v_w = F(p++); bp[i].v_b = F(p++);
    bp[i].po_w = F(p++); bp[i].po_b = F(p++);
    bp[i].ada_self_w = F(p++); bp[i].ada_self_b = F(p++); bp[i].ada_self_g = F(p++);
    bp[i].ada_cross_w = F(p++); bp[i].ada_cross_b = F(p++); bp[i].ada_cross_g = F(p++);
    bp[i].ada_mlp_w = F(p++); bp[i].ada_mlp_b = F(p++); bp[i].ada_mlp_g = F(p++);
    bp[i].mlp_w1 = F(p++); bp[i].mlp_b1 = F(p++); bp[i].mlp_w2 = F(p++); bp[i].mlp_b2 = F(p++);
  }
  const float* fin_g   = F(p++);
  const float* mod_w   = F(p++); const float* mod_b = F(p++);
  const float* proj_w  = F(p++); const float* proj_b = F(p++);

  // ---- workspace layout ----
  char* base = (char*)d_ws;
  float*  ftok    = (float*)base;           // NTOK*PD fp32 (final tokens)
  __bf16* patches = (__bf16*)base;          // NTOK*PD bf16 (embed input; klin/vlin/tmp)
  base += (size_t)NTOK * PD * 4;
  float* xtok = (float*)base;  base += (size_t)NTOK * DD * 4;
  float* ctok = (float*)base;  base += (size_t)NTOK * DD * 4;
  __bf16* qkv  = (__bf16*)base; base += (size_t)NTOK * MMH * 2;   // also MLP hidden
  __bf16* qmod = (__bf16*)base; base += (size_t)NTOK * DD * 2;
  __bf16* obuf = (__bf16*)base; base += (size_t)NTOK * DD * 2;
  __bf16* ctokb= (__bf16*)base; base += (size_t)NTOK * DD * 2;
  __bf16* embf = (__bf16*)base; base += (size_t)BB * 256 * 2;
  __bf16* hid  = (__bf16*)base; base += (size_t)BB * DD * 2;
  float*  temb = (float*)base;  base += (size_t)BB * DD * 4;
  __bf16* stemb= (__bf16*)base; base += (size_t)BB * DD * 2;
  float*  ada  = (float*)base;  base += (size_t)BB * 3 * DD * 4;
  float*  fmod = (float*)base;  base += (size_t)BB * 2 * DD * 4;
  __bf16* wcur = (__bf16*)base;

  __bf16* klin = patches;
  __bf16* vlin = patches + (size_t)NTOK * DD;
  __bf16* tmp  = patches;

  // ---- convert weights fp32 -> bf16 (cheap, then L2-resident) ----
  auto CVT = [&](const float* src, size_t n) -> const __bf16* {
    __bf16* dst = wcur; wcur += n;
    f2bf_kernel<<<(int)((n + 255) / 256), 256, 0, stream>>>(src, dst, (int)n);
    return dst;
  };
  const __bf16* x_emb_wB = CVT(x_emb_w, (size_t)DD * PD);
  const __bf16* c_emb_wB = CVT(c_emb_w, (size_t)DD * PD);
  const __bf16* t_w1B = CVT(t_w1, (size_t)DD * 256);
  const __bf16* t_w2B = CVT(t_w2, (size_t)DD * DD);
  const __bf16* r_w1B = CVT(r_w1, (size_t)DD * 256);
  const __bf16* r_w2B = CVT(r_w2, (size_t)DD * DD);
  const __bf16 *sa_inB[6], *sa_outB[6], *ca_inB[6], *ca_outB[6], *kB[6], *vB[6], *poB[6];
  const __bf16 *adaSB[6], *adaCB[6], *adaMB[6], *mlp1B[6], *mlp2B[6];
  for (int i = 0; i < 6; ++i) {
    sa_inB[i]  = CVT(bp[i].sa_in_w,  (size_t)3 * DD * DD);
    sa_outB[i] = CVT(bp[i].sa_out_w, (size_t)DD * DD);
    adaSB[i]   = CVT(bp[i].ada_self_w, (size_t)3 * DD * DD);
    adaMB[i]   = CVT(bp[i].ada_mlp_w,  (size_t)3 * DD * DD);
    mlp1B[i]   = CVT(bp[i].mlp_w1, (size_t)MMH * DD);
    mlp2B[i]   = CVT(bp[i].mlp_w2, (size_t)DD * MMH);
    if (i < 3) {
      ca_inB[i]  = CVT(bp[i].ca_in_w,  (size_t)3 * DD * DD);
      ca_outB[i] = CVT(bp[i].ca_out_w, (size_t)DD * DD);
      kB[i]      = CVT(bp[i].k_w, (size_t)DD * DD);
      vB[i]      = CVT(bp[i].v_w, (size_t)DD * DD);
      poB[i]     = CVT(bp[i].po_w, (size_t)DD * DD);
      adaCB[i]   = CVT(bp[i].ada_cross_w, (size_t)3 * DD * DD);
    }
  }
  const __bf16* mod_wB  = CVT(mod_w,  (size_t)2 * DD * DD);
  const __bf16* proj_wB = CVT(proj_w, (size_t)PD * DD);

  const int EW = NTOK * DD;

  // ---- token embeddings ----
  patch_gather_kernel<<<(NTOK * PD + 255) / 256, 256, 0, stream>>>(x_in, patches);
  gemm(stream, 0, 0, patches, x_emb_wB, x_emb_b, nullptr, 0, xtok, DD, NTOK, PD, DD);
  addpos_kernel<<<(EW + 255) / 256, 256, 0, stream>>>(xtok, pos_x);
  patch_gather_kernel<<<(NTOK * PD + 255) / 256, 256, 0, stream>>>(cond, patches);
  gemm(stream, 0, 0, patches, c_emb_wB, c_emb_b, nullptr, 0, ctok, DD, NTOK, PD, DD);
  addpos_kernel<<<(EW + 255) / 256, 256, 0, stream>>>(ctok, pos_c);
  f2bf_kernel<<<(EW + 255) / 256, 256, 0, stream>>>(ctok, ctokb, EW);

  // ---- temb = tmlp(t) + tmlp(r); stemb = silu(temb) ----
  tfreq_kernel<<<32, 256, 0, stream>>>(t_in, embf);
  gemm(stream, 1, 1, embf, t_w1B, t_b1, nullptr, 0, hid, DD, BB, 256, DD);
  gemm(stream, 0, 0, hid, t_w2B, t_b2, nullptr, 0, temb, DD, BB, DD, DD);
  tfreq_kernel<<<32, 256, 0, stream>>>(r_in, embf);
  gemm(stream, 1, 1, embf, r_w1B, r_b1, nullptr, 0, hid, DD, BB, 256, DD);
  gemm(stream, 0, 0, hid, r_w2B, r_b2, temb, DD, temb, DD, BB, DD, DD);
  silu_bf_kernel<<<(BB * DD + 255) / 256, 256, 0, stream>>>(temb, stemb, BB * DD);

  for (int i = 0; i < 6; ++i) {
    // --- self attention ---
    gemm(stream, 0, 0, stemb, adaSB[i], bp[i].ada_self_b, nullptr, 0, ada, 3 * DD, BB, DD, 3 * DD);
    adaln_mod_kernel<<<NTOK / 8, 256, 0, stream>>>(xtok, ada, bp[i].ada_self_g, qmod);
    gemm(stream, 0, 1, qmod, sa_inB[i], bp[i].sa_in_b, nullptr, 0, qkv, QKVLD, NTOK, DD, 3 * DD);
    attn_kernel<<<BB * HH * (TT / 64), 128, 0, stream>>>(qkv, obuf);
    gemm(stream, 0, 0, obuf, sa_outB[i], bp[i].sa_out_b, xtok, DD, xtok, DD, NTOK, DD, DD);

    // --- cross attention (first 3 blocks) ---
    if (i < 3) {
      gemm(stream, 0, 0, stemb, adaCB[i], bp[i].ada_cross_b, nullptr, 0, ada, 3 * DD, BB, DD, 3 * DD);
      adaln_mod_kernel<<<NTOK / 8, 256, 0, stream>>>(xtok, ada, bp[i].ada_cross_g, qmod);
      gemm(stream, 0, 1, ctokb, kB[i], bp[i].k_b, nullptr, 0, klin, DD, NTOK, DD, DD);
      gemm(stream, 0, 1, ctokb, vB[i], bp[i].v_b, nullptr, 0, vlin, DD, NTOK, DD, DD);
      // [wq;wk;wv] rows of ca_in -> qkv slices
      gemm(stream, 0, 1, qmod, ca_inB[i],                    bp[i].ca_in_b,          nullptr, 0, qkv,          QKVLD, NTOK, DD, DD);
      gemm(stream, 0, 1, klin, ca_inB[i] + (size_t)DD * DD,  bp[i].ca_in_b + DD,     nullptr, 0, qkv + DD,     QKVLD, NTOK, DD, DD);
      gemm(stream, 0, 1, vlin, ca_inB[i] + (size_t)2*DD*DD,  bp[i].ca_in_b + 2*DD,   nullptr, 0, qkv + 2*DD,   QKVLD, NTOK, DD, DD);
      attn_kernel<<<BB * HH * (TT / 64), 128, 0, stream>>>(qkv, obuf);
      gemm(stream, 0, 1, obuf, ca_outB[i], bp[i].ca_out_b, nullptr, 0, tmp, DD, NTOK, DD, DD);
      gemm(stream, 0, 0, tmp, poB[i], bp[i].po_b, xtok, DD, xtok, DD, NTOK, DD, DD);
    }

    // --- MLP ---
    gemm(stream, 0, 0, stemb, adaMB[i], bp[i].ada_mlp_b, nullptr, 0, ada, 3 * DD, BB, DD, 3 * DD);
    adaln_mod_kernel<<<NTOK / 8, 256, 0, stream>>>(xtok, ada, bp[i].ada_mlp_g, qmod);
    gemm(stream, 2, 1, qmod, mlp1B[i], bp[i].mlp_b1, nullptr, 0, qkv, MMH, NTOK, DD, MMH);
    gemm(stream, 0, 0, qkv, mlp2B[i], bp[i].mlp_b2, xtok, DD, xtok, DD, NTOK, MMH, DD);
  }

  // ---- final layer ----
  gemm(stream, 0, 0, stemb, mod_wB, mod_b, nullptr, 0, fmod, 2 * DD, BB, DD, 2 * DD);
  final_mod_kernel<<<NTOK / 8, 256, 0, stream>>>(xtok, fmod, fin_g, qmod);
  gemm(stream, 0, 0, qmod, proj_wB, proj_b, nullptr, 0, ftok, PD, NTOK, DD, PD);
  unpatchify_kernel<<<(BB * 3 * 256 * 256 + 255) / 256, 256, 0, stream>>>(ftok, (float*)d_out);
}
